// MoeMultiHeadAttention_66726611911083
// MI455X (gfx1250) — matrix operand order repaired and checked by reading, not compiled, not run on hardware
//
#include <hip/hip_runtime.h>
#include <hip/hip_bf16.h>
#include <math.h>

#define B_  32
#define S_  512
#define D_  512
#define E_  8
#define H_  8
#define DH_ 64
#define KT_ 2

typedef __attribute__((ext_vector_type(16))) __bf16 v16bf;
typedef __attribute__((ext_vector_type(8)))  float  v8f;
typedef __attribute__((ext_vector_type(4)))  unsigned int u32x4;
typedef __attribute__((ext_vector_type(4)))  int i32x4;

union Frag16 { v16bf v; u32x4 q[2]; };

__device__ inline v16bf load_frag(const __bf16* p0, const __bf16* p1) {
  Frag16 f;
  f.q[0] = *(const u32x4*)p0;
  f.q[1] = *(const u32x4*)p1;
  return f.v;
}

__device__ inline v8f wmma_bf16(v16bf a, v16bf b, v8f c) {
  // D = A(16x32 bf16) * B(32x16 bf16) + C(f32)
  return __builtin_amdgcn_wmma_f32_16x16x32_bf16(false, a, false, b, (short)0, c, false, false);
}

__device__ inline v8f vzero8() {
  v8f z = {0.f, 0.f, 0.f, 0.f, 0.f, 0.f, 0.f, 0.f};
  return z;
}

// ---------------------------------------------------------------------------
// Async global -> LDS staging (CDNA5 GLOBAL_LOAD_ASYNC_TO_LDS_B128, ASYNCcnt).
// Builtin signature (from hipcc diagnostic): param 0 is `int vector[4]*`.
// Falls back to a plain 128-bit copy if the builtin is unavailable.
// ---------------------------------------------------------------------------
#if defined(__has_builtin) && __has_builtin(__builtin_amdgcn_global_load_async_to_lds_b128)
#define ASYNC_LDS_COPY 1
#else
#define ASYNC_LDS_COPY 0
#endif

__device__ inline void async_copy_b128(const __bf16* gsrc, __bf16* ldst) {
#if ASYNC_LDS_COPY
  __builtin_amdgcn_global_load_async_to_lds_b128(
      (i32x4*)gsrc,
      (__attribute__((address_space(3))) i32x4*)ldst,
      0, 0);
#else
  *(u32x4*)ldst = *(const u32x4*)gsrc;
#endif
}

__device__ inline void async_wait0() {
#if ASYNC_LDS_COPY
#if defined(__has_builtin) && __has_builtin(__builtin_amdgcn_s_wait_asynccnt)
  __builtin_amdgcn_s_wait_asynccnt(0);
#else
  asm volatile("s_wait_asynccnt 0x0" ::: "memory");
#endif
#endif
}

// ---------------------------------------------------------------------------
// Stage 1: xs = sum_s nan_clean(q)   (B x D)
// ---------------------------------------------------------------------------
__global__ void xs_kernel(const float* __restrict__ q, float* __restrict__ xs) {
  int g = blockIdx.x * blockDim.x + threadIdx.x;   // 0 .. B*D
  int b = g / D_, d = g % D_;
  const float* p = q + (size_t)b * S_ * D_ + d;
  float s = 0.f;
  for (int t = 0; t < S_; ++t) {
    float v = p[(size_t)t * D_];
    if (!isnan(v)) s += v;
  }
  xs[g] = s;
}

// ---------------------------------------------------------------------------
// Stage 2a: fp32 -> bf16 conversion of q (nan-clean), k, v
// ---------------------------------------------------------------------------
__global__ void cvt_kernel(const float* __restrict__ q, const float* __restrict__ k,
                           const float* __restrict__ v,
                           __bf16* __restrict__ qb, __bf16* __restrict__ kb,
                           __bf16* __restrict__ vb) {
  size_t g = (size_t)blockIdx.x * blockDim.x + threadIdx.x;
  float qv = q[g];
  if (isnan(qv)) qv = 0.f;
  qb[g] = (__bf16)qv;
  kb[g] = (__bf16)k[g];
  vb[g] = (__bf16)v[g];
}

// ---------------------------------------------------------------------------
// Stage 2b: weights fp32 [e][d][f] -> bf16 transposed [e][f][d] (one-time).
// Transposed layout puts each WMMA B-fragment in two contiguous 16-byte runs,
// so tiles can be staged with async b128 copies and read as ds_load_b128.
// ---------------------------------------------------------------------------
__global__ void cvtw_kernel(const float* __restrict__ Wq, const float* __restrict__ Wk,
                            const float* __restrict__ Wv, const float* __restrict__ Wo,
                            __bf16* __restrict__ Wtq, __bf16* __restrict__ Wtk,
                            __bf16* __restrict__ Wtv, __bf16* __restrict__ Wto) {
  size_t g = (size_t)blockIdx.x * blockDim.x + threadIdx.x;   // 0 .. 4*E*D*D
  const size_t NW = (size_t)E_ * D_ * D_;
  int kind = (int)(g / NW);
  size_t r = g % NW;
  int e = (int)(r / ((size_t)D_ * D_));
  size_t rr = r % ((size_t)D_ * D_);
  int f = (int)(rr / D_);
  int d = (int)(rr % D_);
  const float* src = (kind == 0) ? Wq : (kind == 1) ? Wk : (kind == 2) ? Wv : Wo;
  __bf16* dst = (kind == 0) ? Wtq : (kind == 1) ? Wtk : (kind == 2) ? Wtv : Wto;
  dst[r] = (__bf16)src[(size_t)e * D_ * D_ + (size_t)d * D_ + f];
}

// ---------------------------------------------------------------------------
// Stage 3: gating (noisy top-2), aux loss. One wave; lane = batch row.
// threefry2x32-20 reproducing jax.random.normal(jax.random.key(42), (32,8))
// ---------------------------------------------------------------------------
__device__ inline float tf_normal(int f) {
  const unsigned k0 = 0u, k1 = 42u;
  const unsigned ks2 = k0 ^ k1 ^ 0x1BD11BDAu;
  unsigned ks[3] = {k0, k1, ks2};
  int word = (f < 128) ? 0 : 1;
  unsigned p = (unsigned)(f & 127);
  unsigned x0 = p, x1 = 128u + p;
  x0 += k0; x1 += k1;
  const int R0[4] = {13, 15, 26, 6};
  const int R1[4] = {17, 29, 16, 24};
  for (int g = 0; g < 5; ++g) {
    const int* R = (g & 1) ? R1 : R0;
    for (int r = 0; r < 4; ++r) {
      x0 += x1;
      x1 = (x1 << R[r]) | (x1 >> (32 - R[r]));
      x1 ^= x0;
    }
    x0 += ks[(g + 1) % 3];
    x1 += ks[(g + 2) % 3] + (unsigned)(g + 1);
  }
  unsigned bits = word ? x1 : x0;
  float u01 = __uint_as_float((bits >> 9) | 0x3f800000u) - 1.0f;  // [0,1)
  const float lo = -0.99999994f;                                  // nextafter(-1,0)
  float u = u01 * (1.0f - lo) + lo;
  u = fmaxf(u, lo);
  return 1.4142135623730951f * erfinvf(u);
}

__device__ inline float cv_squared8(const float* x) {
  float m = 0.f;
  for (int e = 0; e < E_; ++e) m += x[e];
  m *= (1.0f / E_);
  float var = 0.f;
  for (int e = 0; e < E_; ++e) { float d = x[e] - m; var += d * d; }
  var *= (1.0f / (E_ - 1));   // ddof=1
  return var / (m * m + 1e-10f);
}

__global__ void gate_kernel(const float* __restrict__ xs,
                            const float* __restrict__ w_gate,
                            const float* __restrict__ w_noise,
                            int* __restrict__ tk_idx, float* __restrict__ tk_gates,
                            float* __restrict__ loss_out) {
  __shared__ float sg[B_ * E_];
  __shared__ float sp[B_ * E_];
  __shared__ float simp[E_], sload[E_];
  int b = threadIdx.x;   // 0..31, one wave

  float clean[E_], noisy[E_], nstd[E_];
  for (int e = 0; e < E_; ++e) {
    float c = 0.f, rn = 0.f;
    for (int d = 0; d < D_; ++d) {
      float x = xs[b * D_ + d];
      c  += x * w_gate[d * E_ + e];
      rn += x * w_noise[d * E_ + e];
    }
    float spl = (rn > 20.f) ? rn : log1pf(expf(rn));
    nstd[e] = spl + 0.01f;
    clean[e] = c;
    noisy[e] = c + tf_normal(b * E_ + e) * nstd[e];
  }

  // top-3 (descending, ties -> lowest index like lax.top_k)
  bool used[E_];
  for (int e = 0; e < E_; ++e) used[e] = false;
  int idx[3]; float val[3];
  for (int t = 0; t < 3; ++t) {
    int best = -1; float bv = -INFINITY;
    for (int e = 0; e < E_; ++e)
      if (!used[e] && noisy[e] > bv) { bv = noisy[e]; best = e; }
    used[best] = true; idx[t] = best; val[t] = bv;
  }

  // softmax over top-2 + GATE_EPS
  float mx = fmaxf(val[0], val[1]);
  float e0 = expf(val[0] - mx), e1 = expf(val[1] - mx);
  float inv = 1.0f / (e0 + e1);
  float g0 = e0 * inv + 1e-9f, g1 = e1 * inv + 1e-9f;
  tk_idx[b * 2 + 0] = idx[0];
  tk_idx[b * 2 + 1] = idx[1];
  tk_gates[b * 2 + 0] = g0;
  tk_gates[b * 2 + 1] = g1;

  float thr_in  = val[2];   // top_logits[:, K_TOP]
  float thr_out = val[1];   // top_logits[:, K_TOP-1]
  for (int e = 0; e < E_; ++e) {
    float g = (e == idx[0]) ? g0 : ((e == idx[1]) ? g1 : 0.f);
    sg[b * E_ + e] = g;
    bool is_in = noisy[e] > thr_in;
    float x = is_in ? (clean[e] - thr_in) / nstd[e] : (clean[e] - thr_out) / nstd[e];
    sp[b * E_ + e] = 0.5f * erfcf(-x * 0.70710678118654752f);   // norm.cdf
  }
  __syncthreads();
  if (threadIdx.x < E_) {
    int e = threadIdx.x;
    float si = 0.f, sl = 0.f;
    for (int bb = 0; bb < B_; ++bb) { si += sg[bb * E_ + e]; sl += sp[bb * E_ + e]; }
    simp[e] = si; sload[e] = sl;
  }
  __syncthreads();
  if (threadIdx.x == 0)
    loss_out[0] = (cv_squared8(simp) + cv_squared8(sload)) * 0.01f;
}

// ---------------------------------------------------------------------------
// Stage 4: expert projections. grid (64, B*KT, 3[q/k/v]); 128 thr (4 waves).
// Each wave: 16 rows x 64 cols via 4 WMMA accumulators; K-loop step 32.
// Pre-transposed bf16 W tile staged via async global->LDS b128 copies.
// ---------------------------------------------------------------------------
__global__ void proj_kernel(const __bf16* __restrict__ qb, const __bf16* __restrict__ kb,
                            const __bf16* __restrict__ vb,
                            const __bf16* __restrict__ Wtq, const float* __restrict__ bq,
                            const __bf16* __restrict__ Wtk, const float* __restrict__ bk,
                            const __bf16* __restrict__ Wtv, const float* __restrict__ bv,
                            const int* __restrict__ tk_idx,
                            __bf16* __restrict__ Qp, __bf16* __restrict__ Kp,
                            __bf16* __restrict__ Vt) {
  __shared__ __bf16 lbT[64 * 32];   // [n][k] tile of W^T
  int kind = blockIdx.z;
  int bj = blockIdx.y;
  int b  = bj >> 1;
  int tx = blockIdx.x & 7, ty = blockIdx.x >> 3;
  int wv = threadIdx.x >> 5, lane = threadIdx.x & 31;
  int e = tk_idx[bj];

  const __bf16* X; const __bf16* Wt; const float* bias;
  if (kind == 0)      { X = qb; Wt = Wtq; bias = bq; }
  else if (kind == 1) { X = kb; Wt = Wtk; bias = bk; }
  else                { X = vb; Wt = Wtv; bias = bv; }

  const __bf16* A   = X  + (size_t)b * S_ * D_;
  const __bf16* WbT = Wt + (size_t)e * D_ * D_;   // [n][k]
  int row0 = ty * 64 + wv * 16;
  int col0 = tx * 64;
  int arow = lane & 15;
  int kbA  = (lane < 16) ? 0 : 8;
  int kbB  = (lane < 16) ? 0 : 16;

  v8f acc[4];
  for (int nt = 0; nt < 4; ++nt) acc[nt] = vzero8();

  for (int k0 = 0; k0 < D_; k0 += 32) {
    __syncthreads();
    // stage 64(n) x 32(k) bf16 tile: 256 x 16B async chunks, 2 per thread
    for (int i = threadIdx.x; i < 256; i += 128) {
      int nn = i >> 2, ko = (i & 3) << 3;
      async_copy_b128(WbT + (size_t)(col0 + nn) * D_ + k0 + ko, &lbT[nn * 32 + ko]);
    }
    async_wait0();
    __syncthreads();
    const __bf16* ap = A + (size_t)(row0 + arow) * D_ + k0 + kbA;
    v16bf af = load_frag(ap, ap + 16);
#pragma unroll
    for (int nt = 0; nt < 4; ++nt) {
      const __bf16* bp = &lbT[(nt * 16 + (lane & 15)) * 32 + kbA];
      v16bf bf = load_frag(bp, bp + 16);
      acc[nt] = wmma_bf16(af, bf, acc[nt]);
    }
  }

#pragma unroll
  for (int nt = 0; nt < 4; ++nt)
#pragma unroll
    for (int r = 0; r < 8; ++r) {
      int row = row0 + r + ((lane < 16) ? 0 : 8);
      int col = col0 + nt * 16 + (lane & 15);
      float vv = acc[nt][r] + bias[e * D_ + col];
      if (kind == 0)
        Qp[(size_t)bj * S_ * D_ + (size_t)row * D_ + col] = (__bf16)vv;
      else if (kind == 1)
        Kp[(size_t)bj * S_ * D_ + (size_t)row * D_ + col] = (__bf16)vv;
      else   // V stored transposed (D x S) for contiguous P@V B-fragments
        Vt[(size_t)bj * D_ * S_ + (size_t)col * S_ + row] = (__bf16)vv;
    }
}

// ---------------------------------------------------------------------------
// Stage 5: attention. One wave per 16 query rows per (b,j,h).
// scores = Q K^T / 8 (+mask) -> LDS f32 -> softmax (all 32 lanes, 2/row)
// -> bf16 P -> ctx = P V.
// ---------------------------------------------------------------------------
__global__ void attn_kernel(const __bf16* __restrict__ Qp, const __bf16* __restrict__ Kp,
                            const __bf16* __restrict__ Vt, const int* __restrict__ mask,
                            __bf16* __restrict__ Ctx) {
  __shared__ float  sS[16 * 512];
  __shared__ __bf16 sP[16 * 512];
  int q0 = blockIdx.x * 16;
  int bj = blockIdx.y;
  int h  = blockIdx.z;
  int lane = threadIdx.x;
  const __bf16* Qb = Qp + (size_t)bj * S_ * D_;
  const __bf16* Kb = Kp + (size_t)bj * S_ * D_;
  const __bf16* Vb = Vt + (size_t)bj * D_ * S_;
  int arow = lane & 15;
  int kbA = (lane < 16) ? 0 : 8;
  int kbB = (lane < 16) ? 0 : 16;

  // preload both Q fragments (dh = 64 -> 2 K-steps of 32)
  v16bf qf[2];
#pragma unroll
  for (int dc = 0; dc < 2; ++dc) {
    const __bf16* qp = Qb + (size_t)(q0 + arow) * D_ + h * DH_ + dc * 32 + kbA;
    qf[dc] = load_frag(qp, qp + 16);
  }

  const float scale = 0.125f;   // 1/sqrt(64)
  for (int ct = 0; ct < S_ / 16; ++ct) {
    v8f acc = vzero8();
#pragma unroll
    for (int dc = 0; dc < 2; ++dc) {
      // B fragment = K^T: column n = key row; d contiguous in memory.
      const __bf16* kp = Kb + (size_t)(ct * 16 + (lane & 15)) * D_ + h * DH_ + dc * 32 + kbA;
      v16bf kf = load_frag(kp, kp + 16);
      acc = wmma_bf16(qf[dc], kf, acc);
    }
#pragma unroll
    for (int r = 0; r < 8; ++r) {
      int m  = r + ((lane < 16) ? 0 : 8);
      int kc = ct * 16 + (lane & 15);
      float v = acc[r] * scale;
      if (mask[(size_t)(q0 + m) * S_ + kc] == 0) v = -1.0e9f;
      sS[m * 512 + kc] = v;
    }
  }
  __syncthreads();

  // softmax: 2 lanes per row, each handling a contiguous 256-column half
  {
    int row = lane >> 1;
    int c0  = (lane & 1) * 256;
    float mx = -INFINITY;
    for (int c = c0; c < c0 + 256; ++c) mx = fmaxf(mx, sS[row * 512 + c]);
    mx = fmaxf(mx, __shfl_xor(mx, 1, 32));
    float sum = 0.f;
    for (int c = c0; c < c0 + 256; ++c) {
      float ev = expf(sS[row * 512 + c] - mx);
      sS[row * 512 + c] = ev;    // lane-private columns: safe overwrite
      sum += ev;
    }
    sum += __shfl_xor(sum, 1, 32);
    float inv = 1.0f / sum;
    for (int c = c0; c < c0 + 256; ++c)
      sP[row * 512 + c] = (__bf16)(sS[row * 512 + c] * inv);
  }
  __syncthreads();

  v8f acc4[4];
  for (int nt = 0; nt < 4; ++nt) acc4[nt] = vzero8();
  for (int k0 = 0; k0 < S_; k0 += 32) {
    const __bf16* pp = &sP[arow * 512 + k0 + kbA];
    v16bf pf = load_frag(pp, pp + 16);
#pragma unroll
    for (int nt = 0; nt < 4; ++nt) {
      int dabs = h * DH_ + nt * 16 + (lane & 15);
      const __bf16* vp = Vb + (size_t)dabs * S_ + k0 + kbA;   // V^T: s contiguous
      v16bf vf = load_frag(vp, vp + 16);
      acc4[nt] = wmma_bf16(pf, vf, acc4[nt]);
    }
  }

  __bf16* Cb = Ctx + (size_t)bj * S_ * D_;
#pragma unroll
  for (int nt = 0; nt < 4; ++nt)
#pragma unroll
    for (int r = 0; r < 8; ++r) {
      int row = q0 + r + ((lane < 16) ? 0 : 8);
      int col = h * DH_ + nt * 16 + (lane & 15);
      Cb[(size_t)row * D_ + col] = (__bf16)acc4[nt][r];
    }
}

// ---------------------------------------------------------------------------
// Stage 6: out_e = ctx @ Wo[e] + bo; combined = sum_j gate_j * exp(out_e);
// out = log(combined==0 ? eps : combined). j-loop in-block -> no atomics.
// ---------------------------------------------------------------------------
__global__ void out_kernel(const __bf16* __restrict__ Ctx, const __bf16* __restrict__ Wto,
                           const float* __restrict__ bo, const int* __restrict__ tk_idx,
                           const float* __restrict__ tk_gates, float* __restrict__ out) {
  __shared__ __bf16 lbT[64 * 32];
  int tx = blockIdx.x & 7, ty = blockIdx.x >> 3;
  int b  = blockIdx.y;
  int wv = threadIdx.x >> 5, lane = threadIdx.x & 31;
  int row0 = ty * 64 + wv * 16;
  int col0 = tx * 64;
  int arow = lane & 15;
  int kbA = (lane < 16) ? 0 : 8;
  int kbB = (lane < 16) ? 0 : 16;

  float comb[4][8];
#pragma unroll
  for (int nt = 0; nt < 4; ++nt)
#pragma unroll
    for (int r = 0; r < 8; ++r) comb[nt][r] = 0.f;

  for (int j = 0; j < KT_; ++j) {
    int bj = b * 2 + j;
    int e = tk_idx[bj];
    float gate = tk_gates[bj];
    const __bf16* A   = Ctx + (size_t)bj * S_ * D_;
    const __bf16* WbT = Wto + (size_t)e * D_ * D_;

    v8f acc[4];
    for (int nt = 0; nt < 4; ++nt) acc[nt] = vzero8();

    for (int k0 = 0; k0 < D_; k0 += 32) {
      __syncthreads();
      for (int i = threadIdx.x; i < 256; i += 128) {
        int nn = i >> 2, ko = (i & 3) << 3;
        async_copy_b128(WbT + (size_t)(col0 + nn) * D_ + k0 + ko, &lbT[nn * 32 + ko]);
      }
      async_wait0();
      __syncthreads();
      const __bf16* ap = A + (size_t)(row0 + arow) * D_ + k0 + kbA;
      v16bf af = load_frag(ap, ap + 16);
#pragma unroll
      for (int nt = 0; nt < 4; ++nt) {
        const __bf16* bp = &lbT[(nt * 16 + (lane & 15)) * 32 + kbA];
        v16bf bf = load_frag(bp, bp + 16);
        acc[nt] = wmma_bf16(af, bf, acc[nt]);
      }
    }
#pragma unroll
    for (int nt = 0; nt < 4; ++nt)
#pragma unroll
      for (int r = 0; r < 8; ++r) {
        int col = col0 + nt * 16 + (lane & 15);
        float vv = acc[nt][r] + bo[e * D_ + col];
        comb[nt][r] += gate * expf(vv);
      }
  }

  const float eps = 2.2204460492503131e-16f;   // np.finfo(float64).eps
#pragma unroll
  for (int nt = 0; nt < 4; ++nt)
#pragma unroll
    for (int r = 0; r < 8; ++r) {
      int row = row0 + r + ((lane < 16) ? 0 : 8);
      int col = col0 + nt * 16 + (lane & 15);
      float v = comb[nt][r];
      if (v == 0.f) v = eps;
      out[((size_t)b * S_ + row) * D_ + col] = logf(v);
    }
}

// ---------------------------------------------------------------------------
extern "C" void kernel_launch(void* const* d_in, const int* in_sizes, int n_in,
                              void* d_out, int out_size, void* d_ws, size_t ws_size,
                              hipStream_t stream) {
  (void)in_sizes; (void)n_in; (void)out_size; (void)ws_size;
  const float* q       = (const float*)d_in[0];
  const float* k       = (const float*)d_in[1];
  const float* v       = (const float*)d_in[2];
  const int*   mask    = (const int*)d_in[3];
  const float* w_gate  = (const float*)d_in[4];
  const float* w_noise = (const float*)d_in[5];
  const float* Wq      = (const float*)d_in[6];
  const float* bq      = (const float*)d_in[7];
  const float* Wk      = (const float*)d_in[8];
  const float* bk      = (const float*)d_in[9];
  const float* Wv      = (const float*)d_in[10];
  const float* bv      = (const float*)d_in[11];
  const float* Wo      = (const float*)d_in[12];
  const float* bo      = (const float*)d_in[13];
  float* out = (float*)d_out;

  char* ws = (char*)d_ws;
  size_t off = 0;
  auto alloc = [&](size_t bytes) { size_t o = off; off += (bytes + 255) & ~(size_t)255; return o; };
  const size_t N  = (size_t)B_ * S_ * D_;
  const size_t NW = (size_t)E_ * D_ * D_;

  float*  xs       = (float*)(ws + alloc((size_t)B_ * D_ * 4));
  int*    tk_idx   = (int*)  (ws + alloc((size_t)B_ * KT_ * 4));
  float*  tk_gates = (float*)(ws + alloc((size_t)B_ * KT_ * 4));
  __bf16* qb       = (__bf16*)(ws + alloc(N * 2));
  __bf16* kb       = (__bf16*)(ws + alloc(N * 2));
  __bf16* vb       = (__bf16*)(ws + alloc(N * 2));
  __bf16* Wtq      = (__bf16*)(ws + alloc(NW * 2));
  __bf16* Wtk      = (__bf16*)(ws + alloc(NW * 2));
  __bf16* Wtv      = (__bf16*)(ws + alloc(NW * 2));
  __bf16* Wto      = (__bf16*)(ws + alloc(NW * 2));
  __bf16* Qp       = (__bf16*)(ws + alloc(N * KT_ * 2));
  __bf16* Kp       = (__bf16*)(ws + alloc(N * KT_ * 2));
  __bf16* Vt       = (__bf16*)(ws + alloc(N * KT_ * 2));
  __bf16* Ctx      = (__bf16*)(ws + alloc(N * KT_ * 2));

  xs_kernel<<<(B_ * D_) / 256, 256, 0, stream>>>(q, xs);
  cvt_kernel<<<(unsigned)(N / 256), 256, 0, stream>>>(q, k, v, qb, kb, vb);
  cvtw_kernel<<<(unsigned)(4 * NW / 256), 256, 0, stream>>>(Wq, Wk, Wv, Wo, Wtq, Wtk, Wtv, Wto);
  gate_kernel<<<1, 32, 0, stream>>>(xs, w_gate, w_noise, tk_idx, tk_gates, out + N);
  proj_kernel<<<dim3(64, B_ * KT_, 3), 128, 0, stream>>>(qb, kb, vb, Wtq, bq, Wtk, bk,
                                                         Wtv, bv, tk_idx, Qp, Kp, Vt);
  attn_kernel<<<dim3(S_ / 16, B_ * KT_, H_), 32, 0, stream>>>(Qp, Kp, Vt, mask, Ctx);
  out_kernel<<<dim3(64, B_), 128, 0, stream>>>(Ctx, Wto, bo, tk_idx, tk_gates, out);
}
